// SelfAttention_v1_9148280340880
// MI455X (gfx1250) — matrix operand order repaired and checked by reading, not compiled, not run on hardware
//
#include <hip/hip_runtime.h>
#include <math.h>

typedef __attribute__((ext_vector_type(16))) _Float16 v16h;
typedef __attribute__((ext_vector_type(8)))  _Float16 v8h;
typedef __attribute__((ext_vector_type(8)))  float    v8f;
typedef __attribute__((ext_vector_type(4)))  float    v4f;

#define N_TOK 8192
#define DIN   256
#define DOUT  128
#define NSPLIT 4

__device__ inline v16h cat16(v8h a, v8h b) {
  v16h r;
#pragma unroll
  for (int i = 0; i < 8; ++i) { r[i] = a[i]; r[i + 8] = b[i]; }
  return r;
}

// ---------------------------------------------------------------------------
// Kernel 1: pack W (f32 [256][128]) -> f16, fragment-major B layout.
// log2(e)/sqrt(128) folded into W_query -> scores land in exp2 domain.
// ---------------------------------------------------------------------------
__global__ void __launch_bounds__(256)
pack_w_kernel(const float* __restrict__ Wq, const float* __restrict__ Wk,
              const float* __restrict__ Wv, _Float16* __restrict__ Wp) {
  int idx = blockIdx.x * 256 + threadIdx.x;          // 0 .. 3*32768-1
  int w    = idx >> 15;
  int r    = idx & 32767;
  int e    = r & 15;
  int lane = (r >> 4) & 31;
  int kc   = (r >> 9) & 7;
  int nt   = (r >> 12) & 7;
  int k    = kc * 32 + ((e >> 3) << 4) + ((lane >> 4) << 3) + (e & 7);
  int col  = nt * 16 + (lane & 15);
  const float* W = (w == 0) ? Wq : (w == 1) ? Wk : Wv;
  float s = (w == 0) ? (0.08838834764831845f * 1.4426950408889634f) : 1.0f;
  Wp[idx] = (_Float16)(W[k * DOUT + col] * s);
}

// ---------------------------------------------------------------------------
// Kernel 2: QKV projection. grid (512, 3), 8 waves/block, wave = col tile.
// ---------------------------------------------------------------------------
__global__ void __launch_bounds__(256)
qkv_kernel(const float* __restrict__ x, const _Float16* __restrict__ Wp,
           _Float16* __restrict__ Qp, _Float16* __restrict__ Kp,
           _Float16* __restrict__ Vp) {
  const int lane = threadIdx.x & 31;
  const int wave = threadIdx.x >> 5;       // col tile 0..7
  const int half = lane >> 4;
  const int nlo  = lane & 15;
  const int rowblk = blockIdx.x;           // 0..511
  const int mat    = blockIdx.y;           // 0..2

  const _Float16* Wm = Wp + mat * 32768;
  _Float16* Out = (mat == 0) ? Qp : (mat == 1) ? Kp : Vp;

  v8f c = {};
#pragma unroll
  for (int kc = 0; kc < 8; ++kc) {
    const float* xr = x + (size_t)(rowblk * 16 + nlo) * DIN + kc * 32;
    v4f a0 = *(const v4f*)(xr + half * 8);
    v4f a1 = *(const v4f*)(xr + half * 8 + 4);
    v4f a2 = *(const v4f*)(xr + 16 + half * 8);
    v4f a3 = *(const v4f*)(xr + 16 + half * 8 + 4);
    v16h a;
#pragma unroll
    for (int i = 0; i < 4; ++i) {
      a[i]      = (_Float16)a0[i];
      a[i + 4]  = (_Float16)a1[i];
      a[i + 8]  = (_Float16)a2[i];
      a[i + 12] = (_Float16)a3[i];
    }
    const _Float16* wp = Wm + ((wave * 8 + kc) * 32 + lane) * 16;
    v16h b = cat16(*(const v8h*)wp, *(const v8h*)(wp + 8));
    c = __builtin_amdgcn_wmma_f32_16x16x32_f16(false, a, false, b,
                                               (short)0, c, false, false);
  }
#pragma unroll
  for (int v = 0; v < 8; ++v) {
    int m = half * 8 + v;
    Out[(size_t)(rowblk * 16 + m) * DOUT + wave * 16 + nlo] = (_Float16)c[v];
  }
}

// ---------------------------------------------------------------------------
// Kernel 3: flash attention over a key range, LDS double-buffered.
// FINAL=true : single split, writes normalized f32 output directly.
// FINAL=false: writes unnormalized partial O + per-row (m,l) (exp2 domain).
// ---------------------------------------------------------------------------
template <bool FINAL>
__global__ void __launch_bounds__(256)
flash_kernel(const _Float16* __restrict__ Qp, const _Float16* __restrict__ Kp,
             const _Float16* __restrict__ Vp, float* __restrict__ obuf,
             float* __restrict__ mlbuf, int tiles_per_split) {
  __shared__ _Float16 Kl[2][32 * 136];    // K tile, padded row stride 136
  __shared__ _Float16 Vt[2][128 * 40];    // V^T tile, padded row stride 40
  __shared__ _Float16 Pl[8][16 * 40];     // per-wave P tile (C->A relayout)

  const int tid  = threadIdx.x;
  const int lane = tid & 31;
  const int wave = tid >> 5;
  const int half = lane >> 4;
  const int nlo  = lane & 15;
  const int qbase = blockIdx.x * 128 + wave * 16;
  const int split = blockIdx.y;
  const int t0 = split * tiles_per_split;

  // Q A-fragments (row-major f16, pre-scaled by log2e/sqrt(dk) via Wq)
  v16h qa[4];
#pragma unroll
  for (int c = 0; c < 4; ++c) {
    const _Float16* p = Qp + (size_t)(qbase + nlo) * DOUT + c * 32 + half * 16;
    qa[c] = cat16(*(const v8h*)p, *(const v8h*)(p + 8));
  }

  v8f o[8];
#pragma unroll
  for (int nt = 0; nt < 8; ++nt) o[nt] = (v8f){};
  float mrow[8], lrow[8];
#pragma unroll
  for (int v = 0; v < 8; ++v) { mrow[v] = -INFINITY; lrow[v] = 0.f; }

  const int key0 = tid >> 3;            // 0..31
  const int dseg = (tid & 7) * 16;      // 0..112

  // preload tile t0 into buffer 0
  {
    const _Float16* kg = Kp + (size_t)(t0 * 32 + key0) * DOUT + dseg;
    v8h k0 = *(const v8h*)kg, k1 = *(const v8h*)(kg + 8);
    const _Float16* vg = Vp + (size_t)(t0 * 32 + key0) * DOUT + dseg;
    v8h v0 = *(const v8h*)vg, v1 = *(const v8h*)(vg + 8);
    _Float16* kd = &Kl[0][key0 * 136 + dseg];
    *(v8h*)kd = k0; *(v8h*)(kd + 8) = k1;
#pragma unroll
    for (int i = 0; i < 8; ++i) {
      Vt[0][(dseg + i) * 40 + key0]     = v0[i];
      Vt[0][(dseg + 8 + i) * 40 + key0] = v1[i];
    }
  }

  for (int t = 0; t < tiles_per_split; ++t) {
    const int cur = t & 1;
    __syncthreads();   // buffer `cur` ready; buffer `1-cur` free

    // issue global loads for tile t+1 (fly under this iteration's compute)
    v8h k0n = {}, k1n = {}, v0n = {}, v1n = {};
    const bool havenext = (t + 1 < tiles_per_split);
    if (havenext) {
      const _Float16* kg = Kp + (size_t)((t0 + t + 1) * 32 + key0) * DOUT + dseg;
      k0n = *(const v8h*)kg; k1n = *(const v8h*)(kg + 8);
      const _Float16* vg = Vp + (size_t)((t0 + t + 1) * 32 + key0) * DOUT + dseg;
      v0n = *(const v8h*)vg; v1n = *(const v8h*)(vg + 8);
    }
    if (t + 2 < tiles_per_split) {  // pull tile t+2 toward caches
      __builtin_prefetch(Kp + (size_t)((t0 + t + 2) * 32 + key0) * DOUT + dseg, 0, 1);
      __builtin_prefetch(Vp + (size_t)((t0 + t + 2) * 32 + key0) * DOUT + dseg, 0, 1);
    }

    // ----- scores: two 16-key groups, 4 k-chunks each -----
    v8f s0 = (v8f){}, s1 = (v8f){};
#pragma unroll
    for (int c = 0; c < 4; ++c) {
      const _Float16* kb0 = &Kl[cur][nlo * 136 + c * 32 + half * 16];
      const _Float16* kb1 = &Kl[cur][(16 + nlo) * 136 + c * 32 + half * 16];
      v16h b0 = cat16(*(const v8h*)kb0, *(const v8h*)(kb0 + 8));
      v16h b1 = cat16(*(const v8h*)kb1, *(const v8h*)(kb1 + 8));
      s0 = __builtin_amdgcn_wmma_f32_16x16x32_f16(false, qa[c], false, b0,
                                                  (short)0, s0, false, false);
      s1 = __builtin_amdgcn_wmma_f32_16x16x32_f16(false, qa[c], false, b1,
                                                  (short)0, s1, false, false);
    }

    // ----- online softmax (exp2 domain) -----
    _Float16* Pw = Pl[wave];
#pragma unroll
    for (int v = 0; v < 8; ++v) {
      float xm = fmaxf(s0[v], s1[v]);
#pragma unroll
      for (int mk = 1; mk < 16; mk <<= 1) xm = fmaxf(xm, __shfl_xor(xm, mk, 16));
      float mn = fmaxf(mrow[v], xm);
      float sc = exp2f(mrow[v] - mn);
      float p0 = exp2f(s0[v] - mn);
      float p1 = exp2f(s1[v] - mn);
      float rs = p0 + p1;
#pragma unroll
      for (int mk = 1; mk < 16; mk <<= 1) rs += __shfl_xor(rs, mk, 16);
      lrow[v] = lrow[v] * sc + rs;
      mrow[v] = mn;
#pragma unroll
      for (int nt = 0; nt < 8; ++nt) o[nt][v] *= sc;
      int m = half * 8 + v;
      Pw[m * 40 + nlo]      = (_Float16)p0;
      Pw[m * 40 + 16 + nlo] = (_Float16)p1;
    }
    __builtin_amdgcn_wave_barrier();
    asm volatile("s_wait_dscnt 0x0" ::: "memory");  // per-wave LDS RAW fence
    __builtin_amdgcn_wave_barrier();

    // ----- O += P @ V -----
    const _Float16* pr = Pw + nlo * 40 + half * 16;
    v16h pa = cat16(*(const v8h*)pr, *(const v8h*)(pr + 8));
#pragma unroll
    for (int nt = 0; nt < 8; ++nt) {
      const _Float16* vb = &Vt[cur][(nt * 16 + nlo) * 40 + half * 16];
      v16h b = cat16(*(const v8h*)vb, *(const v8h*)(vb + 8));
      o[nt] = __builtin_amdgcn_wmma_f32_16x16x32_f16(false, pa, false, b,
                                                     (short)0, o[nt], false, false);
    }

    // stage tile t+1 into the other buffer
    if (havenext) {
      _Float16* kd = &Kl[1 - cur][key0 * 136 + dseg];
      *(v8h*)kd = k0n; *(v8h*)(kd + 8) = k1n;
#pragma unroll
      for (int i = 0; i < 8; ++i) {
        Vt[1 - cur][(dseg + i) * 40 + key0]     = v0n[i];
        Vt[1 - cur][(dseg + 8 + i) * 40 + key0] = v1n[i];
      }
    }
  }

  // ----- epilogue -----
  if (FINAL) {
#pragma unroll
    for (int v = 0; v < 8; ++v) {
      float inv = 1.0f / lrow[v];
      int m = half * 8 + v;
      float* orow = obuf + (size_t)(qbase + m) * DOUT;
#pragma unroll
      for (int nt = 0; nt < 8; ++nt) orow[nt * 16 + nlo] = o[nt][v] * inv;
    }
  } else {
#pragma unroll
    for (int v = 0; v < 8; ++v) {
      int m = half * 8 + v;
      size_t prow = (size_t)split * N_TOK + qbase + m;
      float* orow = obuf + prow * DOUT;
#pragma unroll
      for (int nt = 0; nt < 8; ++nt) orow[nt * 16 + nlo] = o[nt][v];
      if (nlo == v) {                 // one lane per row writes (m,l)
        mlbuf[prow * 2]     = mrow[v];
        mlbuf[prow * 2 + 1] = lrow[v];
      }
    }
  }
}

// ---------------------------------------------------------------------------
// Kernel 4: merge split-K partials (exp2 domain).
// ---------------------------------------------------------------------------
__global__ void __launch_bounds__(256)
merge_kernel(const float* __restrict__ Opart, const float* __restrict__ mlbuf,
             float* __restrict__ out) {
  int idx = blockIdx.x * 256 + threadIdx.x;   // row*128 + col
  int row = idx >> 7;
  int col = idx & 127;
  float mstar = -INFINITY;
#pragma unroll
  for (int s = 0; s < NSPLIT; ++s)
    mstar = fmaxf(mstar, mlbuf[((size_t)s * N_TOK + row) * 2]);
  float lsum = 0.f, osum = 0.f;
#pragma unroll
  for (int s = 0; s < NSPLIT; ++s) {
    size_t prow = (size_t)s * N_TOK + row;
    float w = exp2f(mlbuf[prow * 2] - mstar);
    lsum += mlbuf[prow * 2 + 1] * w;
    osum += Opart[prow * DOUT + col] * w;
  }
  out[idx] = osum / lsum;
}

// ---------------------------------------------------------------------------
extern "C" void kernel_launch(void* const* d_in, const int* in_sizes, int n_in,
                              void* d_out, int out_size, void* d_ws, size_t ws_size,
                              hipStream_t stream) {
  const float* x  = (const float*)d_in[0];
  const float* Wq = (const float*)d_in[1];
  const float* Wk = (const float*)d_in[2];
  const float* Wv = (const float*)d_in[3];
  float* out = (float*)d_out;

  char* ws = (char*)d_ws;
  const size_t off_Wp = 0;                       // 196608 B
  const size_t off_Qp = 196608;                  // 2 MB
  const size_t off_Kp = off_Qp + 2097152;        // 2 MB
  const size_t off_Vp = off_Kp + 2097152;        // 2 MB
  const size_t off_Op = off_Vp + 2097152;        // NSPLIT * 4 MB partial O
  const size_t off_ml = off_Op + (size_t)NSPLIT * N_TOK * DOUT * 4;
  const size_t need_split = off_ml + (size_t)NSPLIT * N_TOK * 2 * 4;

  _Float16* Wp = (_Float16*)(ws + off_Wp);
  _Float16* Qp = (_Float16*)(ws + off_Qp);
  _Float16* Kp = (_Float16*)(ws + off_Kp);
  _Float16* Vp = (_Float16*)(ws + off_Vp);

  pack_w_kernel<<<384, 256, 0, stream>>>(Wq, Wk, Wv, Wp);
  qkv_kernel<<<dim3(512, 3), 256, 0, stream>>>(x, Wp, Qp, Kp, Vp);

  if (ws_size >= need_split) {
    float* Opart = (float*)(ws + off_Op);
    float* mlbuf = (float*)(ws + off_ml);
    flash_kernel<false><<<dim3(64, NSPLIT), 256, 0, stream>>>(
        Qp, Kp, Vp, Opart, mlbuf, 256 / NSPLIT);
    merge_kernel<<<(N_TOK * DOUT) / 256, 256, 0, stream>>>(Opart, mlbuf, out);
  } else {
    flash_kernel<true><<<dim3(64, 1), 256, 0, stream>>>(
        Qp, Kp, Vp, out, nullptr, 256);
  }
}
